// SequencePenaltyCriterion_82179904242090
// MI455X (gfx1250) — compile-verified, matrix operand order checked by reading, not run on hardware
//
#include <hip/hip_runtime.h>
#include <math.h>

// Problem shape fixed by the reference setup_inputs(): pred is (B, V, S) f32.
#define B_     8
#define V_     32000
#define S_     512
#define NGRAM  4
#define P_     (S_ - NGRAM)      // 508 four-gram start positions
#define CHUNKS 128               // V split for parallelism: 250 v per chunk
#define VPC    (V_ / CHUNKS)     // 250
#define NTILE  (S_ / 16)         // 32 row tiles of 16 for WMMA

typedef __attribute__((ext_vector_type(8))) int   v8i;
typedef __attribute__((ext_vector_type(4))) float v4f;

// ---------------------------------------------------------------------------
// Kernel 1: streaming online-softmax partials (HBM-bound, coalesced 128-bit
// non-temporal loads: 524 MB single pass, never re-read -> TH=NT so the
// stream does not thrash L2/WGP$; partial outputs stay RT so k2 hits L2).
// Block = 256 threads: lower 128 lanes = s/4 position, upper bit = v-halfchunk.
// Each thread keeps 4 independent (max, sum, argmax) states.
// ---------------------------------------------------------------------------
__global__ __launch_bounds__(256) void k1_partial(const float* __restrict__ pred,
                                                  float* __restrict__ pm,
                                                  float* __restrict__ ps,
                                                  int*   __restrict__ pa) {
  const int b     = blockIdx.y;
  const int chunk = blockIdx.x * 2 + (threadIdx.x >> 7);
  const int t     = threadIdx.x & 127;
  const int s0    = t * 4;
  const int v0    = chunk * VPC;
  const float* base = pred + ((size_t)(b * V_ + v0)) * S_ + s0;

  float m0 = -INFINITY, m1 = -INFINITY, m2 = -INFINITY, m3 = -INFINITY;
  float q0 = 0.f, q1 = 0.f, q2 = 0.f, q3 = 0.f;
  int   a0 = 0, a1 = 0, a2 = 0, a3 = 0;

  for (int v = 0; v < VPC; ++v) {
    v4f x = __builtin_nontemporal_load((const v4f*)(base + (size_t)v * S_));
    const int vi = v0 + v;
    if (x.x > m0) { q0 = q0 * __expf(m0 - x.x) + 1.0f; m0 = x.x; a0 = vi; }
    else          { q0 += __expf(x.x - m0); }
    if (x.y > m1) { q1 = q1 * __expf(m1 - x.y) + 1.0f; m1 = x.y; a1 = vi; }
    else          { q1 += __expf(x.y - m1); }
    if (x.z > m2) { q2 = q2 * __expf(m2 - x.z) + 1.0f; m2 = x.z; a2 = vi; }
    else          { q2 += __expf(x.z - m2); }
    if (x.w > m3) { q3 = q3 * __expf(m3 - x.w) + 1.0f; m3 = x.w; a3 = vi; }
    else          { q3 += __expf(x.w - m3); }
  }

  const int p = (b * CHUNKS + chunk) * S_ + s0;
  *(float4*)(pm + p) = make_float4(m0, m1, m2, m3);
  *(float4*)(ps + p) = make_float4(q0, q1, q2, q3);
  *(int4*)  (pa + p) = make_int4(a0, a1, a2, a3);
}

// ---------------------------------------------------------------------------
// Kernel 2: merge partials per (b,s) -> token + penalty; zero rep[] and out.
// ---------------------------------------------------------------------------
__global__ __launch_bounds__(256) void k2_finalize(const float* __restrict__ pm,
                                                   const float* __restrict__ ps,
                                                   const int*   __restrict__ pa,
                                                   int*   __restrict__ toks,
                                                   float* __restrict__ pen,
                                                   int*   __restrict__ rep,
                                                   float* __restrict__ out) {
  const int idx = blockIdx.x * blockDim.x + threadIdx.x;   // 0 .. B*S-1
  if (idx == 0) out[0] = 0.0f;
  const int b = idx / S_;
  const int s = idx - b * S_;

  float M = -INFINITY, SU = 0.0f;
  int   A = 0;
  for (int c = 0; c < CHUNKS; ++c) {
    const int p = (b * CHUNKS + c) * S_ + s;
    const float m  = pm[p];
    const float su = ps[p];
    const int   a  = pa[p];
    if (m > M) { SU = SU * __expf(M - m) + su; M = m; A = a; }  // strict >: first-occurrence argmax
    else       { SU += su * __expf(m - M); }
  }
  // lprob(argmax) = -log(SU); prob = 1/SU
  float om = 1.0f - 1.0f / SU;
  om = fmaxf(om, 1e-20f);
  toks[idx] = A;
  pen[idx]  = -__logf(om);
  rep[idx]  = 0;
}

// ---------------------------------------------------------------------------
// Kernel 3: 4-gram repeat detection via V_WMMA_I32_16X16X64_IU8.
// Each 4-gram -> 8 bytes (tokens < 32768 => 2 bytes each).
// dist^2(i,j) = |gi|^2 + |gj|^2 - 2*gi.gj == 0  <=>  4-grams equal.
// One wave per 16-row tile mi; loop lower-triangular tiles ni <= mi.
// A-tile norms (nA per result slot) are loop-invariant -> hoisted shuffles.
// ---------------------------------------------------------------------------
__global__ __launch_bounds__(32) void k3_ngram_wmma(const int* __restrict__ toks,
                                                    int* __restrict__ rep) {
  const int b    = blockIdx.y;
  const int mi   = blockIdx.x;          // 0..NTILE-1
  const int lane = threadIdx.x;         // one wave32
  const int* trow = toks + b * S_;
  const int half = lane & 15;

  // Build A fragment (rows mi*16 .. mi*16+15).
  const int rowA = mi * 16 + half;
  unsigned tA0, tA1, tA2, tA3;
  if (rowA < P_) {
    tA0 = (unsigned)trow[rowA];     tA1 = (unsigned)trow[rowA + 1];
    tA2 = (unsigned)trow[rowA + 2]; tA3 = (unsigned)trow[rowA + 3];
  } else {
    tA0 = tA1 = tA2 = tA3 = 0xFFFFu;  // sentinel: hi byte 255 > any real token's hi byte
  }
  const unsigned pA0 = (tA0 & 0xFFFFu) | ((tA1 & 0xFFFFu) << 16);
  const unsigned pA1 = (tA2 & 0xFFFFu) | ((tA3 & 0xFFFFu) << 16);
  int normA = 0;
  {
    unsigned tt[4] = {tA0, tA1, tA2, tA3};
#pragma unroll
    for (int k = 0; k < 4; ++k) {
      const int lo = (int)(tt[k] & 0xFFu);
      const int hi = (int)((tt[k] >> 8) & 0xFFu);
      normA += lo * lo + hi * hi;
    }
  }
  v8i afr = {};
  afr[0] = (lane < 16) ? (int)pA0 : 0;
  afr[1] = (lane < 16) ? (int)pA1 : 0;

  // Hoisted: per result slot r this lane reports M = r + (lane>=16 ? 8 : 0);
  // normA[M] lives in lane M (< 16). Invariant across the ni loop.
  int nAr[8];
#pragma unroll
  for (int r = 0; r < 8; ++r) {
    const int m_loc = r + ((lane >> 4) << 3);
    nAr[r] = __shfl(normA, m_loc, 32);
  }

  for (int ni = 0; ni <= mi; ++ni) {
    const int rowB = ni * 16 + half;
    unsigned tB0, tB1, tB2, tB3;
    if (rowB < P_) {
      tB0 = (unsigned)trow[rowB];     tB1 = (unsigned)trow[rowB + 1];
      tB2 = (unsigned)trow[rowB + 2]; tB3 = (unsigned)trow[rowB + 3];
    } else {
      tB0 = tB1 = tB2 = tB3 = 0xFFFFu;
    }
    const unsigned pB0 = (tB0 & 0xFFFFu) | ((tB1 & 0xFFFFu) << 16);
    const unsigned pB1 = (tB2 & 0xFFFFu) | ((tB3 & 0xFFFFu) << 16);
    int normB = 0;
    {
      unsigned tt[4] = {tB0, tB1, tB2, tB3};
#pragma unroll
      for (int k = 0; k < 4; ++k) {
        const int lo = (int)(tt[k] & 0xFFu);
        const int hi = (int)((tt[k] >> 8) & 0xFFu);
        normB += lo * lo + hi * hi;
      }
    }
    v8i bfr = {};
    bfr[0] = (lane < 16) ? (int)pB0 : 0;
    bfr[1] = (lane < 16) ? (int)pB1 : 0;

    v8i c = {};
    // (sgn_a, A, sgn_b, B, C, reuse_a, reuse_b) -> v_wmma_i32_16x16x64_iu8
    c = __builtin_amdgcn_wmma_i32_16x16x64_iu8(false, afr, false, bfr, c, false, false);

    // normB for column N = lane&15 lives in lane (lane&15): 1 bpermute per tile.
    const int nB = __shfl(normB, half, 32);
#pragma unroll
    for (int r = 0; r < 8; ++r) {
      const int m_loc = r + ((lane >> 4) << 3);       // C/D layout: lanes 16-31 hold M=r+8
      const int gi    = mi * 16 + m_loc;
      const int gj    = ni * 16 + half;
      const int dist  = nAr[r] + nB - 2 * c[r];
      if (dist == 0 && gj < gi && gi < P_ && gj < P_) {
        rep[b * S_ + gi] = 1;                          // benign race: all writers store 1
      }
    }
  }
}

// ---------------------------------------------------------------------------
// Kernel 4: dilate repeat flags by NGRAM, masked sum of penalties, /B.
// One block per batch row, 512 threads.
// ---------------------------------------------------------------------------
__global__ __launch_bounds__(512) void k4_reduce(const float* __restrict__ pen,
                                                 const int*   __restrict__ rep,
                                                 float* __restrict__ out) {
  const int b = blockIdx.x;
  const int s = threadIdx.x;
  bool masked = false;
#pragma unroll
  for (int k = 0; k < NGRAM; ++k) {
    const int i = s - k;
    if (i >= 0 && i < P_ && rep[b * S_ + i] != 0) masked = true;
  }
  float val = masked ? pen[b * S_ + s] : 0.0f;

  __shared__ float red[16];
#pragma unroll
  for (int off = 16; off > 0; off >>= 1) val += __shfl_down(val, off, 32);
  if ((s & 31) == 0) red[s >> 5] = val;
  __syncthreads();
  if (s < 16) {
    float v = red[s];
#pragma unroll
    for (int off = 8; off > 0; off >>= 1) v += __shfl_down(v, off, 32);
    if (s == 0) atomicAdd(out, v * (1.0f / (float)B_));
  }
}

// ---------------------------------------------------------------------------
extern "C" void kernel_launch(void* const* d_in, const int* in_sizes, int n_in,
                              void* d_out, int out_size, void* d_ws, size_t ws_size,
                              hipStream_t stream) {
  (void)in_sizes; (void)n_in; (void)out_size; (void)ws_size;
  const float* pred = (const float*)d_in[0];
  float* out = (float*)d_out;

  char* ws = (char*)d_ws;
  const size_t np = (size_t)B_ * CHUNKS * S_;        // 524288 partial states
  float* pm   = (float*)ws;  ws += np * sizeof(float);
  float* psum = (float*)ws;  ws += np * sizeof(float);
  int*   pa   = (int*)ws;    ws += np * sizeof(int);
  int*   toks = (int*)ws;    ws += (size_t)B_ * S_ * sizeof(int);
  float* pen  = (float*)ws;  ws += (size_t)B_ * S_ * sizeof(float);
  int*   rep  = (int*)ws;    ws += (size_t)B_ * S_ * sizeof(int);

  dim3 g1(CHUNKS / 2, B_);
  k1_partial<<<g1, 256, 0, stream>>>(pred, pm, psum, pa);

  k2_finalize<<<(B_ * S_) / 256, 256, 0, stream>>>(pm, psum, pa, toks, pen, rep, out);

  dim3 g3(NTILE, B_);
  k3_ngram_wmma<<<g3, 32, 0, stream>>>(toks, rep);

  k4_reduce<<<B_, S_, 0, stream>>>(pen, rep, out);
}
